// MorphModel_21921513079318
// MI455X (gfx1250) — matrix operand order repaired and checked by reading, not compiled, not run on hardware
//
#include <hip/hip_runtime.h>
#include <hip/hip_bf16.h>
#include <cstdint>
#include <cstddef>

// MoE FFN for MI455X (gfx1250, wave32, WMMA).
// Dense-per-expert (matches reference): out = sum_e we[:,e] * (relu(x@W1e+b1e)@W2e+b2e)
// ~1.1 TFLOP dense vs ~320MB HBM traffic @23.3TB/s -> compute bound -> bf16 WMMA (f32 acc).
// Workspace: we[B*E] f32 (256KB) + Ht[HID*B] bf16 (64MB).

typedef __attribute__((ext_vector_type(16))) __bf16 v16bf;
typedef __attribute__((ext_vector_type(8)))  float v8f;

constexpr int NB = 8192, NIN = 1024, NHID = 4096, NOUT = 1024, NE = 8;
constexpr int TM = 128, TN = 128, TK = 64;   // macro tile; LDS = 2*(128*64 + 128*64)*2B = 64KB

union Frag { uint4 q[2]; v16bf v; };

__device__ __forceinline__ unsigned cvt2_bf16(float a, float b) {
  unsigned ua = __builtin_bit_cast(unsigned, a);
  unsigned ub = __builtin_bit_cast(unsigned, b);
  ua += 0x7FFFu + ((ua >> 16) & 1u);   // round-to-nearest-even
  ub += 0x7FFFu + ((ub >> 16) & 1u);
  return (ua >> 16) | (ub & 0xFFFF0000u);
}

// ---------------- Gating: logits -> softmax -> top2 -> renormalized we[B][E] ----
__global__ __launch_bounds__(256) void gate_kernel(const float* __restrict__ x,
                                                   const float* __restrict__ Wg,
                                                   const float* __restrict__ bg,
                                                   float* __restrict__ we) {
  __shared__ float red[256][NE];
  const int b = blockIdx.x, tid = threadIdx.x;
  float p[NE];
#pragma unroll
  for (int e = 0; e < NE; ++e) p[e] = 0.0f;
  for (int i = tid; i < NIN; i += 256) {
    const float xv = x[(size_t)b * NIN + i];
    const float* wr = Wg + (size_t)i * NE;
#pragma unroll
    for (int e = 0; e < NE; ++e) p[e] += xv * wr[e];
  }
#pragma unroll
  for (int e = 0; e < NE; ++e) red[tid][e] = p[e];
  __syncthreads();
  for (int s = 128; s > 0; s >>= 1) {
    if (tid < s) {
#pragma unroll
      for (int e = 0; e < NE; ++e) red[tid][e] += red[tid + s][e];
    }
    __syncthreads();
  }
  if (tid == 0) {
    float lg[NE];
#pragma unroll
    for (int e = 0; e < NE; ++e) lg[e] = red[0][e] + bg[e];
    int i1 = 0; float l1 = lg[0];
#pragma unroll
    for (int e = 1; e < NE; ++e) if (lg[e] > l1) { l1 = lg[e]; i1 = e; }
    int i2 = -1; float l2 = -3.4e38f;
#pragma unroll
    for (int e = 0; e < NE; ++e) if (e != i1 && lg[e] > l2) { l2 = lg[e]; i2 = e; }
    const float r = __expf(l2 - l1);        // p2/p1; renormalized top-2 weights
    const float inv = 1.0f / (1.0f + r);
#pragma unroll
    for (int e = 0; e < NE; ++e)
      we[(size_t)b * NE + e] = (e == i1) ? inv : ((e == i2) ? r * inv : 0.0f);
  }
}

// ---------------- GEMM1: Ht[n][m] = bf16(relu(X@W1e + b1e))  (transposed store) --
__global__ __launch_bounds__(256) void ffn1_kernel(const float* __restrict__ X,
                                                   const float* __restrict__ W1e,
                                                   const float* __restrict__ b1e,
                                                   unsigned short* __restrict__ Ht) {
  __shared__ unsigned short As[2][TM * TK];
  __shared__ unsigned short Bs[2][TN * TK];
  const int tid = threadIdx.x;
  const int nb = blockIdx.x * TN, mb = blockIdx.y * TM;
  const int lane = tid & 31, wid = tid >> 5;
  const int wm = (wid & 1) * 64, wn = (wid >> 1) * 32;
  const int l15 = lane & 15, kg = lane >> 4;
  const int arow0 = tid >> 4, ac4 = tid & 15;   // A: 128 rows x 16 float4
  const int bkp0 = tid >> 5, bc4 = tid & 31;    // B: 32 k-pairs x 32 float4

  v8f acc[4][2] = {};

  auto load_tile = [&](int buf, int k0) {
#pragma unroll
    for (int r = 0; r < 8; ++r) {               // A row-major: b64 packed stores
      const int row = arow0 + r * 16;
      const float4 v = *(const float4*)(X + (size_t)(mb + row) * NIN + k0 + ac4 * 4);
      uint2 pk; pk.x = cvt2_bf16(v.x, v.y); pk.y = cvt2_bf16(v.z, v.w);
      *(uint2*)&As[buf][row * TK + ac4 * 4] = pk;
      if (k0 + 2 * TK < NIN)
        __builtin_prefetch(X + (size_t)(mb + row) * NIN + k0 + 2 * TK + ac4 * 4, 0, 0);
    }
#pragma unroll
    for (int r = 0; r < 4; ++r) {               // B transposed: two k-rows -> b32 stores
      const int k = (bkp0 + r * 8) * 2;
      const float4 v0 = *(const float4*)(W1e + (size_t)(k0 + k) * NHID + nb + bc4 * 4);
      const float4 v1 = *(const float4*)(W1e + (size_t)(k0 + k + 1) * NHID + nb + bc4 * 4);
      const int n0 = bc4 * 4;                   // Bs[n][k..k+1] packed pair
      *(unsigned*)&Bs[buf][(n0 + 0) * TK + k] = cvt2_bf16(v0.x, v1.x);
      *(unsigned*)&Bs[buf][(n0 + 1) * TK + k] = cvt2_bf16(v0.y, v1.y);
      *(unsigned*)&Bs[buf][(n0 + 2) * TK + k] = cvt2_bf16(v0.z, v1.z);
      *(unsigned*)&Bs[buf][(n0 + 3) * TK + k] = cvt2_bf16(v0.w, v1.w);
      if (k0 + 2 * TK < NIN)
        __builtin_prefetch(W1e + (size_t)(k0 + k + 2 * TK) * NHID + nb + bc4 * 4, 0, 0);
    }
  };

  auto compute = [&](int buf) {
#pragma unroll
    for (int ks = 0; ks < TK; ks += 32) {
      Frag a[4], b[2];
#pragma unroll
      for (int mi = 0; mi < 4; ++mi) {          // A frag: row=m, halves0-7 k=kg*8.., 8-15 k=+16
        const unsigned short* p = &As[buf][(wm + mi * 16 + l15) * TK + ks + kg * 8];
        a[mi].q[0] = *(const uint4*)p;
        a[mi].q[1] = *(const uint4*)(p + 16);
      }
#pragma unroll
      for (int ni = 0; ni < 2; ++ni) {          // B frag: col=n, halves0-7 k=kg*8.., 8-15 k=+16
        const unsigned short* p = &Bs[buf][(wn + ni * 16 + l15) * TK + ks + kg * 8];
        b[ni].q[0] = *(const uint4*)p;
        b[ni].q[1] = *(const uint4*)(p + 16);
      }
#pragma unroll
      for (int mi = 0; mi < 4; ++mi)
#pragma unroll
        for (int ni = 0; ni < 2; ++ni)
          acc[mi][ni] = __builtin_amdgcn_wmma_f32_16x16x32_bf16(
              false, a[mi].v, false, b[ni].v, (short)0, acc[mi][ni], false, false);
    }
  };

  load_tile(0, 0);
  __syncthreads();
  constexpr int NKT = NIN / TK;
#pragma unroll 1
  for (int kt = 0; kt < NKT; ++kt) {
    const int cur = kt & 1;
    if (kt + 1 < NKT) load_tile(cur ^ 1, (kt + 1) * TK);
    compute(cur);
    __syncthreads();
  }

  // Epilogue: relu(+bias) -> bf16, transposed store Ht[n][m0..m0+7] (one 16B store)
#pragma unroll
  for (int mi = 0; mi < 4; ++mi) {
    const int m0 = mb + wm + mi * 16 + kg * 8;
#pragma unroll
    for (int ni = 0; ni < 2; ++ni) {
      const int n = nb + wn + ni * 16 + l15;
      const float bias = b1e[n];
      float t[8];
#pragma unroll
      for (int i = 0; i < 8; ++i) {
        const float v = acc[mi][ni][i] + bias;
        t[i] = v > 0.0f ? v : 0.0f;
      }
      uint4 q;
      q.x = cvt2_bf16(t[0], t[1]); q.y = cvt2_bf16(t[2], t[3]);
      q.z = cvt2_bf16(t[4], t[5]); q.w = cvt2_bf16(t[6], t[7]);
      *(uint4*)(Ht + (size_t)n * NB + m0) = q;
    }
  }
}

// ---------------- GEMM2: Out (+)= we[:,e] * (h@W2e + b2e), h read k-major -------
__global__ __launch_bounds__(256) void ffn2_kernel(const unsigned short* __restrict__ Ht,
                                                   const float* __restrict__ W2e,
                                                   const float* __restrict__ b2e,
                                                   const float* __restrict__ Wgt,
                                                   const int e, float* __restrict__ Out,
                                                   const int first) {
  __shared__ unsigned short As[2][TM * TK];
  __shared__ unsigned short Bs[2][TN * TK];
  const int tid = threadIdx.x;
  const int nb = blockIdx.x * TN, mb = blockIdx.y * TM;
  const int lane = tid & 31, wid = tid >> 5;
  const int wm = (wid & 1) * 64, wn = (wid >> 1) * 32;
  const int l15 = lane & 15, kg = lane >> 4;
  const int akp0 = tid >> 5, ac = tid & 31;   // A: 32 k-pairs x 32 m-chunks (4 bf16)
  const int bkp0 = tid >> 5, bc4 = tid & 31;  // B: 32 k-pairs x 32 float4

  v8f acc[4][2] = {};

  auto load_tile = [&](int buf, int k0) {
#pragma unroll
    for (int r = 0; r < 4; ++r) {             // Ht[k][m] bf16: two k-rows -> b32 packed stores
      const int k = (akp0 + r * 8) * 2;
      const uint2 h0 = *(const uint2*)(Ht + (size_t)(k0 + k) * NB + mb + ac * 4);
      const uint2 h1 = *(const uint2*)(Ht + (size_t)(k0 + k + 1) * NB + mb + ac * 4);
      const int m0l = ac * 4;                 // As[m][k..k+1] = (h(k,m), h(k+1,m))
      *(unsigned*)&As[buf][(m0l + 0) * TK + k] = (h0.x & 0xFFFFu)  | (h1.x << 16);
      *(unsigned*)&As[buf][(m0l + 1) * TK + k] = (h0.x >> 16)      | (h1.x & 0xFFFF0000u);
      *(unsigned*)&As[buf][(m0l + 2) * TK + k] = (h0.y & 0xFFFFu)  | (h1.y << 16);
      *(unsigned*)&As[buf][(m0l + 3) * TK + k] = (h0.y >> 16)      | (h1.y & 0xFFFF0000u);
      if (k0 + 2 * TK < NHID)
        __builtin_prefetch(Ht + (size_t)(k0 + k + 2 * TK) * NB + mb + ac * 4, 0, 0);
    }
#pragma unroll
    for (int r = 0; r < 4; ++r) {             // W2 transposed: two k-rows -> b32 stores
      const int k = (bkp0 + r * 8) * 2;
      const float4 v0 = *(const float4*)(W2e + (size_t)(k0 + k) * NOUT + nb + bc4 * 4);
      const float4 v1 = *(const float4*)(W2e + (size_t)(k0 + k + 1) * NOUT + nb + bc4 * 4);
      const int n0 = bc4 * 4;
      *(unsigned*)&Bs[buf][(n0 + 0) * TK + k] = cvt2_bf16(v0.x, v1.x);
      *(unsigned*)&Bs[buf][(n0 + 1) * TK + k] = cvt2_bf16(v0.y, v1.y);
      *(unsigned*)&Bs[buf][(n0 + 2) * TK + k] = cvt2_bf16(v0.z, v1.z);
      *(unsigned*)&Bs[buf][(n0 + 3) * TK + k] = cvt2_bf16(v0.w, v1.w);
      if (k0 + 2 * TK < NHID)
        __builtin_prefetch(W2e + (size_t)(k0 + k + 2 * TK) * NOUT + nb + bc4 * 4, 0, 0);
    }
  };

  auto compute = [&](int buf) {
#pragma unroll
    for (int ks = 0; ks < TK; ks += 32) {
      Frag a[4], b[2];
#pragma unroll
      for (int mi = 0; mi < 4; ++mi) {
        const unsigned short* p = &As[buf][(wm + mi * 16 + l15) * TK + ks + kg * 8];
        a[mi].q[0] = *(const uint4*)p;
        a[mi].q[1] = *(const uint4*)(p + 16);
      }
#pragma unroll
      for (int ni = 0; ni < 2; ++ni) {
        const unsigned short* p = &Bs[buf][(wn + ni * 16 + l15) * TK + ks + kg * 8];
        b[ni].q[0] = *(const uint4*)p;
        b[ni].q[1] = *(const uint4*)(p + 16);
      }
#pragma unroll
      for (int mi = 0; mi < 4; ++mi)
#pragma unroll
        for (int ni = 0; ni < 2; ++ni)
          acc[mi][ni] = __builtin_amdgcn_wmma_f32_16x16x32_bf16(
              false, a[mi].v, false, b[ni].v, (short)0, acc[mi][ni], false, false);
    }
  };

  load_tile(0, 0);
  __syncthreads();
  constexpr int NKT = NHID / TK;
#pragma unroll 1
  for (int kt = 0; kt < NKT; ++kt) {
    const int cur = kt & 1;
    if (kt + 1 < NKT) load_tile(cur ^ 1, (kt + 1) * TK);
    compute(cur);
    __syncthreads();
  }

  // Epilogue: (+bias) * we[m,e], accumulate into Out (blocks own disjoint tiles)
#pragma unroll
  for (int mi = 0; mi < 4; ++mi) {
    const int m0 = mb + wm + mi * 16 + kg * 8;
    float wrow[8];
#pragma unroll
    for (int i = 0; i < 8; ++i) wrow[i] = Wgt[(size_t)(m0 + i) * NE + e];
#pragma unroll
    for (int ni = 0; ni < 2; ++ni) {
      const int n = nb + wn + ni * 16 + l15;
      const float bias = b2e[n];
#pragma unroll
      for (int i = 0; i < 8; ++i) {
        const float val = (acc[mi][ni][i] + bias) * wrow[i];
        float* po = Out + (size_t)(m0 + i) * NOUT + n;
        if (first) *po = val;
        else *po = *po + val;
      }
    }
  }
}

extern "C" void kernel_launch(void* const* d_in, const int* in_sizes, int n_in,
                              void* d_out, int out_size, void* d_ws, size_t ws_size,
                              hipStream_t stream) {
  (void)in_sizes; (void)n_in; (void)out_size; (void)ws_size;
  const float* x  = (const float*)d_in[0];
  const float* Wg = (const float*)d_in[1];
  const float* bg = (const float*)d_in[2];
  const float* W1 = (const float*)d_in[3];
  const float* b1 = (const float*)d_in[4];
  const float* W2 = (const float*)d_in[5];
  const float* b2 = (const float*)d_in[6];
  float* out = (float*)d_out;
  float* we = (float*)d_ws;                                             // [B, E] f32
  unsigned short* Ht =
      (unsigned short*)((char*)d_ws + (size_t)NB * NE * sizeof(float)); // [HID, B] bf16

  gate_kernel<<<NB, 256, 0, stream>>>(x, Wg, bg, we);
  for (int e = 0; e < NE; ++e) {
    ffn1_kernel<<<dim3(NHID / TN, NB / TM), 256, 0, stream>>>(
        x, W1 + (size_t)e * NIN * NHID, b1 + (size_t)e * NHID, Ht);
    ffn2_kernel<<<dim3(NOUT / TN, NB / TM), 256, 0, stream>>>(
        Ht, W2 + (size_t)e * NHID * NOUT, b2 + (size_t)e * NOUT, we, e, out, e == 0 ? 1 : 0);
  }
}
